// S6Operator_78580721647779
// MI455X (gfx1250) — matrix operand order repaired and checked by reading, not compiled, not run on hardware
//
#include <hip/hip_runtime.h>
#include <hip/hip_bf16.h>

#define DM   2048
#define DI   4096
#define DS   16
#define DTR  128
#define LSEQ 1024
#define XPN  (DTR + 2 * DS)   // 160

typedef __attribute__((ext_vector_type(16))) __bf16         v16bf;
typedef __attribute__((ext_vector_type(16))) unsigned short v16us;
typedef __attribute__((ext_vector_type(8)))  float          v8f;
typedef __attribute__((ext_vector_type(4)))  float          v4f;
typedef __attribute__((ext_vector_type(4)))  unsigned int   v4u;
typedef __attribute__((ext_vector_type(2)))  __bf16         v2bf;

__device__ __forceinline__ unsigned short f2bf(float f) {
  unsigned int u = __builtin_bit_cast(unsigned int, f);
  u += 0x7fffu + ((u >> 16) & 1u);   // round-to-nearest-even
  return (unsigned short)(u >> 16);
}

// two f32 -> packed bf16; native v_cvt_pk_bf16_f32 when the builtin exists
__device__ __forceinline__ unsigned int pack2bf(float lo, float hi) {
#if __has_builtin(__builtin_amdgcn_cvt_pk_bf16_f32)
  v2bf p = __builtin_amdgcn_cvt_pk_bf16_f32(lo, hi);
  return __builtin_bit_cast(unsigned int, p);
#else
  return (unsigned int)f2bf(lo) | ((unsigned int)f2bf(hi) << 16);
#endif
}

__device__ __forceinline__ float silu_f(float s) {
  return s / (1.0f + __expf(-s));
}

// CDNA5 async copy: 16B global -> LDS per lane, tracked by ASYNCcnt.
// Generic pointers to __shared__ carry the LDS byte offset in their low 32
// bits on amdgcn, so truncation yields the VDST LDS-address operand.
__device__ __forceinline__ void async_g2l_b128(const void* g, const void* l) {
  unsigned int loff = (unsigned int)(unsigned long long)l;
  asm volatile("global_load_async_to_lds_b128 %0, %1, off"
               :: "v"(loff), "v"(g) : "memory");
}
__device__ __forceinline__ void wait_async0() {
  asm volatile("s_wait_asynccnt 0" ::: "memory");
}

// ---------------------------------------------------------------------------
// RMSNorm: one block per row of x (1024 x 2048), output bf16 bits
// ---------------------------------------------------------------------------
__global__ void rmsnorm_kernel(const float* __restrict__ x,
                               const float* __restrict__ w,
                               unsigned short* __restrict__ xn) {
  int row = blockIdx.x;
  const float* xr = x + row * DM;
  float s = 0.0f;
  for (int i = threadIdx.x; i < DM; i += 256) { float v = xr[i]; s += v * v; }
  for (int o = 16; o; o >>= 1) s += __shfl_xor(s, o, 32);
  __shared__ float red[8];
  __shared__ float scale;
  if ((threadIdx.x & 31) == 0) red[threadIdx.x >> 5] = s;
  __syncthreads();
  if (threadIdx.x == 0) {
    float t = 0.0f;
    for (int i = 0; i < 8; i++) t += red[i];
    scale = rsqrtf(t / (float)DM + 1e-5f);
  }
  __syncthreads();
  for (int i = threadIdx.x; i < DM; i += 256)
    xn[row * DM + i] = f2bf(xr[i] * scale * w[i]);
}

// ---------------------------------------------------------------------------
// Tiled WMMA GEMM with double-buffered LDS + software pipelining.
//   C[M,N] = epi( A[M,K] * B[K,N] )
//   A bf16 path: staged with GLOBAL_LOAD_ASYNC_TO_LDS_B128 (ASYNCcnt).
//   A fp32 path: prefetched to regs, converted, stored next to B staging.
//   B: fp32 -> bf16 while staging, stored transposed [n][K] in LDS.
//   sA rows store K in natural order (contiguous k = 16h+s, matching B) so each
//   lane's fragment is one contiguous 32B LDS read.
//   Block tile 64(M) x 128(N), 8 waves, wave tile 32x32 (2x2 WMMA).
//   EPI: 0 plain, 1 += extra[row*ldc+col], 2 softplus(acc+extra[col])
// ---------------------------------------------------------------------------
template <bool A_BF16, int EPI>
__global__ __launch_bounds__(256) void gemm_wmma(
    const void* __restrict__ Ap, const float* __restrict__ B,
    float* __restrict__ C, const float* __restrict__ extra,
    int M, int N, int K, int lda, int ldb, int ldc) {
  __shared__ alignas(16) unsigned short sA[2][64 * 32];   // [buf][row][K]
  __shared__ alignas(16) unsigned short sBT[2][128 * 32]; // [buf][n][K]

  const unsigned short* Abf = (const unsigned short*)Ap;
  const float*          Af  = (const float*)Ap;

  int tid  = threadIdx.x;
  int lane = tid & 31;
  int wv   = tid >> 5;
  int wm   = (wv >> 2) * 32;
  int wn   = (wv & 3) * 32;
  int row0 = blockIdx.y * 64;
  int col0 = blockIdx.x * 128;

  v8f acc[2][2];
  for (int i = 0; i < 2; i++)
    for (int j = 0; j < 2; j++)
      for (int v = 0; v < 8; v++) acc[i][j][v] = 0.0f;

  // A staging: thread -> (row, 8-wide K block), blocks in natural K order
  int ar_  = tid >> 2;
  int ab   = tid & 3;
  int apos = ab * 8;
  // fragment read indices
  int am    = lane & 15;
  int ahalf = (lane >> 4) * 16;
  int bn    = lane & 15;
  int bkh   = (lane >> 4) * 16;

  // prefetch registers
  v4f pa0, pa1;        // fp32-A path
  v4f pb0[2], pb1[2];  // B path (2 tasks/thread)
  for (int i = 0; i < 4; i++) { pa0[i] = 0.0f; pa1[i] = 0.0f; }

  auto issueA = [&](int k0, int buf) {
    if (A_BF16) {
      async_g2l_b128(&Abf[(row0 + ar_) * lda + k0 + ab * 8],
                     &sA[buf][ar_ * 32 + apos]);
    } else {
      const float* ap = &Af[(row0 + ar_) * lda + k0 + ab * 8];
      pa0 = *(const v4f*)ap;
      pa1 = *(const v4f*)(ap + 4);
    }
  };
  auto loadB = [&](int k0) {
#pragma unroll
    for (int rep = 0; rep < 2; rep++) {
      int task = tid + rep * 256;
      int rp   = task >> 5;
      int cg   = task & 31;
      int gc   = col0 + cg * 4;
      if (gc < N) {                      // N is a multiple of 4
        pb0[rep] = *(const v4f*)&B[(k0 + 2 * rp) * ldb + gc];
        pb1[rep] = *(const v4f*)&B[(k0 + 2 * rp + 1) * ldb + gc];
      } else {
        for (int i = 0; i < 4; i++) { pb0[rep][i] = 0.0f; pb1[rep][i] = 0.0f; }
      }
    }
  };
  auto storeStaged = [&](int buf) {
    if (!A_BF16) {
      v4u aw;
      aw[0] = pack2bf(pa0[0], pa0[1]);
      aw[1] = pack2bf(pa0[2], pa0[3]);
      aw[2] = pack2bf(pa1[0], pa1[1]);
      aw[3] = pack2bf(pa1[2], pa1[3]);
      *(v4u*)&sA[buf][ar_ * 32 + apos] = aw;
    }
#pragma unroll
    for (int rep = 0; rep < 2; rep++) {
      int task = tid + rep * 256;
      int rp   = task >> 5;
      int cg   = task & 31;
#pragma unroll
      for (int i = 0; i < 4; i++)
        *(unsigned int*)&sBT[buf][(cg * 4 + i) * 32 + 2 * rp] =
            pack2bf(pb0[rep][i], pb1[rep][i]);
    }
  };

  const int KT = K >> 5;

  // prologue: stage tile 0 into buffer 0
  issueA(0, 0);
  loadB(0);
  storeStaged(0);
  if (A_BF16) wait_async0();
  __syncthreads();

  int cur = 0;
  for (int kt = 0; kt < KT; kt++) {
    int  nxt  = cur ^ 1;
    bool more = (kt + 1 < KT);
    // prefetch next tile (async A goes straight to LDS buf[nxt])
    if (more) { issueA((kt + 1) << 5, nxt); loadB((kt + 1) << 5); }

    // fragments: one contiguous 32B LDS read each
    v16bf afr[2], bfr[2];
#pragma unroll
    for (int im = 0; im < 2; im++)
      afr[im] = __builtin_bit_cast(
          v16bf, *(const v16us*)&sA[cur][(wm + im * 16 + am) * 32 + ahalf]);
#pragma unroll
    for (int in = 0; in < 2; in++)
      bfr[in] = __builtin_bit_cast(
          v16bf, *(const v16us*)&sBT[cur][(wn + in * 16 + bn) * 32 + bkh]);

#pragma unroll
    for (int im = 0; im < 2; im++)
#pragma unroll
      for (int in = 0; in < 2; in++)
        acc[im][in] = __builtin_amdgcn_wmma_f32_16x16x32_bf16(
            false, afr[im], false, bfr[in], (short)0, acc[im][in], false, false);

    if (more) {
      storeStaged(nxt);
      if (A_BF16) wait_async0();
    }
    __syncthreads();
    cur = nxt;
  }

  // epilogue: C/D layout — lane half selects M base (0/8), VGPR v is M within
  int mh = (lane >> 4) * 8;
  int nn = lane & 15;
#pragma unroll
  for (int im = 0; im < 2; im++) {
#pragma unroll
    for (int in = 0; in < 2; in++) {
      int gcol = col0 + wn + in * 16 + nn;
      if (gcol >= N) continue;
#pragma unroll
      for (int v = 0; v < 8; v++) {
        int grow  = row0 + wm + im * 16 + mh + v;
        float val = acc[im][in][v];
        if (EPI == 1) val += extra[grow * ldc + gcol];
        if (EPI == 2) {
          val += extra[gcol];
          val = (val > 20.0f) ? val : log1pf(__expf(val));
        }
        C[grow * ldc + gcol] = val;
      }
    }
  }
}

// ---------------------------------------------------------------------------
// Depthwise causal conv (k=4) + SiLU on xs half; SiLU on res half
// ---------------------------------------------------------------------------
__global__ void conv_silu_kernel(const float* __restrict__ xz,
                                 const float* __restrict__ cw,
                                 const float* __restrict__ cb,
                                 float* __restrict__ u,
                                 unsigned short* __restrict__ xs_act,
                                 float* __restrict__ sres) {
  int idx = blockIdx.x * 256 + threadIdx.x;  // [0, LSEQ*DI)
  int t = idx >> 12;
  int d = idx & (DI - 1);
  float acc = cb[d];
#pragma unroll
  for (int j = 0; j < 4; j++) {
    int tt = t - 3 + j;
    if (tt >= 0) acc += cw[j * DI + d] * xz[tt * (2 * DI) + d];
  }
  float s = silu_f(acc);
  u[idx] = s;
  xs_act[idx] = f2bf(s);
  sres[idx] = silu_f(xz[t * (2 * DI) + DI + d]);
}

// ---------------------------------------------------------------------------
// Selective scan: one lane per channel d, 16 states in registers.
// B/C (32 floats, shared by all channels) are loaded one-per-lane and
// broadcast with wave32 lane shuffles -- no LDS, no barriers on the
// sequential critical path. Fuses y*silu(res) -> bf16 output.
// ---------------------------------------------------------------------------
__global__ __launch_bounds__(256) void scan_kernel(
    const float* __restrict__ delta, const float* __restrict__ u,
    const float* __restrict__ xdbc, const float* __restrict__ A_log,
    const float* __restrict__ Dp, const float* __restrict__ sres,
    unsigned short* __restrict__ ybf) {
  int d = blockIdx.x * 256 + threadIdx.x;
  int lane = threadIdx.x & 31;
  float A[DS], h[DS];
#pragma unroll
  for (int n = 0; n < DS; n++) {
    A[n] = -__expf(A_log[d * DS + n]);
    h[n] = 0.0f;
  }
  float Dd = Dp[d];
  for (int t = 0; t < LSEQ; t++) {
    float bcv = xdbc[t * XPN + DTR + lane];  // lane n: B[n]; lane 16+n: C[n]
    float dlt = delta[t * DI + d];
    float ut  = u[t * DI + d];
    float y = 0.0f;
#pragma unroll
    for (int n = 0; n < DS; n++) {
      float Bn = __shfl(bcv, n, 32);
      float Cn = __shfl(bcv, DS + n, 32);
      float dA = __expf(dlt * A[n]);
      h[n] = h[n] * dA + (dlt * Bn) * ut;
      y += h[n] * Cn;
    }
    y += ut * Dd;
    ybf[t * DI + d] = f2bf(y * sres[t * DI + d]);
  }
}

// ---------------------------------------------------------------------------
extern "C" void kernel_launch(void* const* d_in, const int* in_sizes, int n_in,
                              void* d_out, int out_size, void* d_ws, size_t ws_size,
                              hipStream_t stream) {
  const float* x      = (const float*)d_in[0];
  const float* w_norm = (const float*)d_in[1];
  const float* W_in   = (const float*)d_in[2];
  const float* conv_w = (const float*)d_in[3];
  const float* conv_b = (const float*)d_in[4];
  const float* W_xp   = (const float*)d_in[5];
  const float* W_dt   = (const float*)d_in[6];
  const float* b_dt   = (const float*)d_in[7];
  const float* A_log  = (const float*)d_in[8];
  const float* Dvec   = (const float*)d_in[9];
  const float* W_out  = (const float*)d_in[10];
  float* out = (float*)d_out;

  char* ws = (char*)d_ws;
  // layout (bytes); delta/ybf overlay the dead xz region
  unsigned short* xn    = (unsigned short*)(ws + 0);                    //  4 MB
  float*          xz    = (float*)(ws + (4u << 20));                    // 32 MB
  float*          delta = (float*)(ws + (4u << 20));                    // 16 MB (overlay)
  unsigned short* ybf   = (unsigned short*)(ws + (20u << 20));          //  8 MB (overlay)
  float*          u     = (float*)(ws + (37u << 20));                   // 16 MB
  unsigned short* xsact = (unsigned short*)(ws + (53u << 20));          //  8 MB
  float*          sres  = (float*)(ws + (61u << 20));                   // 16 MB
  float*          xdbc  = (float*)(ws + (77u << 20));                   // <1 MB

  // 1) RMSNorm -> bf16 xn
  rmsnorm_kernel<<<LSEQ, 256, 0, stream>>>(x, w_norm, xn);

  // 2) xz = xn @ W_in   (1024 x 2048 x 8192)
  gemm_wmma<true, 0><<<dim3((2 * DI + 127) / 128, LSEQ / 64), 256, 0, stream>>>(
      xn, W_in, xz, nullptr, LSEQ, 2 * DI, DM, DM, 2 * DI, 2 * DI);

  // 3) depthwise causal conv + SiLU, silu(res)
  conv_silu_kernel<<<(LSEQ * DI) / 256, 256, 0, stream>>>(
      xz, conv_w, conv_b, u, xsact, sres);

  // 4) xdbc = xs_act @ W_xp   (1024 x 4096 x 160)
  gemm_wmma<true, 0><<<dim3((XPN + 127) / 128, LSEQ / 64), 256, 0, stream>>>(
      xsact, W_xp, xdbc, nullptr, LSEQ, XPN, DI, DI, XPN, XPN);

  // 5) delta = softplus(dt_r @ W_dt + b_dt)  (1024 x 128 x 4096), fused epilogue
  gemm_wmma<false, 2><<<dim3((DI + 127) / 128, LSEQ / 64), 256, 0, stream>>>(
      xdbc, W_dt, delta, b_dt, LSEQ, DI, DTR, XPN, DI, DI);

  // 6) selective scan, fused * silu(res) -> bf16
  scan_kernel<<<DI / 256, 256, 0, stream>>>(delta, u, xdbc, A_log, Dvec, sres, ybf);

  // 7) out = ybf @ W_out + x   (1024 x 4096 x 2048), fused residual
  gemm_wmma<true, 1><<<dim3((DM + 127) / 128, LSEQ / 64), 256, 0, stream>>>(
      ybf, W_out, out, x, LSEQ, DM, DI, DI, DM, DM);
}
